// CrossModalityAttention_51479478010163
// MI455X (gfx1250) — matrix operand order repaired and checked by reading, not compiled, not run on hardware
//
#include <hip/hip_runtime.h>
#include <cstdint>

#define B_  8
#define L_  2048
#define D_  512
#define BL_ (B_*L_)

typedef __bf16 v16bf __attribute__((ext_vector_type(16)));
typedef float  v8f   __attribute__((ext_vector_type(8)));
typedef unsigned int v4u __attribute__((ext_vector_type(4)));

union Frag { v16bf v; v4u u[2]; };

__device__ __forceinline__ unsigned short f2bf(float f) {
  unsigned int u = __float_as_uint(f);
  u += 0x7FFFu + ((u >> 16) & 1u);   // round-to-nearest-even
  return (unsigned short)(u >> 16);
}

__device__ __forceinline__ v8f wmma_bf16(const Frag& a, const Frag& b, v8f c) {
  return __builtin_amdgcn_wmma_f32_16x16x32_bf16(false, a.v, false, b.v,
                                                 (short)0, c, false, false);
}

// ---------------- prep: fp32 -> bf16 activations ----------------
__global__ void cvt_inputs(const float* __restrict__ raw, const float* __restrict__ hand,
                           unsigned short* __restrict__ xr, unsigned short* __restrict__ xh) {
  size_t i = (size_t)blockIdx.x * blockDim.x + threadIdx.x;
  xr[i] = f2bf(raw[i]);
  xh[i] = f2bf(hand[i]);
}

// ---------------- prep: W -> W^T bf16  (Wt[p][n][k] = W_p[k][n]) ----------------
__global__ void cvt_weights(const float* w0, const float* w1, const float* w2,
                            const float* w3, const float* w4, const float* w5,
                            unsigned short* __restrict__ wt) {
  int i = blockIdx.x * blockDim.x + threadIdx.x;     // p*D*D + n*D + k
  int p = i / (D_ * D_);
  int r = i % (D_ * D_);
  int n = r / D_, k = r % D_;
  const float* w = (p == 0) ? w0 : (p == 1) ? w1 : (p == 2) ? w2
                  : (p == 3) ? w3 : (p == 4) ? w4 : w5;
  wt[i] = f2bf(w[k * D_ + n]);
}

// ---------------- projections: W-slice resident in LDS, 16x64 tile per wave ----------------
// grid: 6 proj * 8 Nblocks * 128 Mchunks; block = 8 waves; block tile = 128 x 64
#define WLPAD 520   // 512 + 8 : row stride = 260 dwords = 4 banks offset per row
__global__ void __launch_bounds__(256) proj_kernel(
    const unsigned short* __restrict__ xr, const unsigned short* __restrict__ xh,
    const unsigned short* __restrict__ wt,
    const float* b0, const float* b1, const float* b2,
    const float* b3, const float* b4, const float* b5,
    unsigned short* __restrict__ qr, unsigned short* __restrict__ kr,
    unsigned short* __restrict__ qh, unsigned short* __restrict__ kh,
    unsigned short* __restrict__ vtr, unsigned short* __restrict__ vth) {
  __shared__ __align__(16) unsigned short Wl[64][WLPAD];

  int bx   = blockIdx.x;
  int p    = bx >> 10;                 // / (8*128)
  int rem  = bx & 1023;
  int nblk = rem >> 7;                 // 0..7
  int mch  = rem & 127;                // 0..127
  int tid  = threadIdx.x;
  int wv   = tid >> 5, lane = tid & 31, m = lane & 15, hi = lane >> 4;
  int N0   = nblk * 64;
  int M0   = mch * 128 + wv * 16;

  const unsigned short* X = (p < 3) ? xr : xh;
  const unsigned short* W = wt + (size_t)p * D_ * D_;
  const float* bias = (p == 0) ? b0 : (p == 1) ? b1 : (p == 2) ? b2
                     : (p == 3) ? b3 : (p == 4) ? b4 : b5;

  // stage 64 x 512 bf16 W^T slice into LDS (each thread: 16 x 16B)
  for (int i = tid; i < 4096; i += 256) {
    int n = i >> 6;
    int kc = (i & 63) * 8;
    *(v4u*)&Wl[n][kc] = *(const v4u*)(W + (size_t)(N0 + n) * D_ + kc);
  }
  __syncthreads();

  const unsigned short* xrow = X + (size_t)(M0 + m) * D_;

  v8f z = {};
  v8f c[4]; c[0] = z; c[1] = z; c[2] = z; c[3] = z;
  #pragma unroll
  for (int kk = 0; kk < 16; ++kk) {
    int d0 = kk * 32;
    Frag a;
    a.u[0] = *(const v4u*)(xrow + d0 + hi * 8);
    a.u[1] = *(const v4u*)(xrow + d0 + 16 + hi * 8);
    #pragma unroll
    for (int nt = 0; nt < 4; ++nt) {
      Frag b;
      b.u[0] = *(const v4u*)&Wl[nt * 16 + m][d0 + hi * 8];
      b.u[1] = *(const v4u*)&Wl[nt * 16 + m][d0 + 16 + hi * 8];
      c[nt] = wmma_bf16(a, b, c[nt]);
    }
  }

  if ((p % 3) < 2) {  // Q/K: row-major [BL, D] bf16
    unsigned short* O = (p == 0) ? qr : (p == 1) ? kr : (p == 3) ? qh : kh;
    #pragma unroll
    for (int nt = 0; nt < 4; ++nt) {
      float bv = bias[N0 + nt * 16 + m];
      #pragma unroll
      for (int r = 0; r < 8; ++r) {
        int row = M0 + hi * 8 + r;
        O[(size_t)row * D_ + N0 + nt * 16 + m] = f2bf(c[nt][r] + bv);
      }
    }
  } else {            // V: transposed [B][D][L] bf16
    unsigned short* O = (p == 2) ? vtr : vth;
    #pragma unroll
    for (int nt = 0; nt < 4; ++nt) {
      float bv = bias[N0 + nt * 16 + m];
      int d = N0 + nt * 16 + m;
      #pragma unroll
      for (int r = 0; r < 8; ++r) {
        int row = M0 + hi * 8 + r;           // = b*L + l
        int bb = row >> 11, l = row & (L_ - 1);
        O[((size_t)bb * D_ + d) * L_ + l] = f2bf(c[nt][r] + bv);
      }
    }
  }
}

// ---------------- flash attention: 32 queries / block, 8 waves split D ----------------
// grid: (L/32, B, 2); block = 256
#define SPPAD 36
__global__ void __launch_bounds__(256) attn_kernel(
    const unsigned short* __restrict__ qrp, const unsigned short* __restrict__ khp,
    const unsigned short* __restrict__ vtrp, const unsigned short* __restrict__ qhp,
    const unsigned short* __restrict__ krp, const unsigned short* __restrict__ vthp,
    float* __restrict__ outp) {
  __shared__ float Sp[8][32][SPPAD];       // per-wave partial scores (padded)
  __shared__ __align__(16) unsigned short P[32][40];   // probs bf16 (padded)
  __shared__ float mrun[32], lrun[32], rscale[32];

  int z = blockIdx.z;
  const unsigned short* Q  = z ? qhp : qrp;
  const unsigned short* K  = z ? krp : khp;
  const unsigned short* Vt = z ? vthp : vtrp;
  float* Out = outp + (size_t)z * BL_ * D_;

  int tid = threadIdx.x;
  int w = tid >> 5, lane = tid & 31, m = lane & 15, hi = lane >> 4;
  int b = blockIdx.y, m0 = blockIdx.x * 32;
  int dsl = w * 64;                        // this wave's D-slice

  if (tid < 32) { mrun[tid] = -3.0e38f; lrun[tid] = 0.f; }
  __syncthreads();

  // Q fragments: 2 M-tiles x 2 k-steps over this wave's D-slice
  Frag qf[2][2];
  #pragma unroll
  for (int mt = 0; mt < 2; ++mt) {
    const unsigned short* qrow = Q + (size_t)(b * L_ + m0 + mt * 16 + m) * D_ + dsl;
    #pragma unroll
    for (int kk = 0; kk < 2; ++kk) {
      qf[mt][kk].u[0] = *(const v4u*)(qrow + kk * 32 + hi * 8);
      qf[mt][kk].u[1] = *(const v4u*)(qrow + kk * 32 + 16 + hi * 8);
    }
  }

  v8f zv = {};
  v8f o[2][4];
  #pragma unroll
  for (int mt = 0; mt < 2; ++mt)
    #pragma unroll
    for (int nt = 0; nt < 4; ++nt) o[mt][nt] = zv;

  int row8 = tid >> 3, c8 = tid & 7;       // softmax mapping: 8 threads per row

  for (int kt = 0; kt < L_ / 32; ++kt) {
    int k0 = kt * 32;
    // partial S = Q_slice * K_slice^T  (K fragment reused across both M-tiles)
    v8f s[2][2];
    #pragma unroll
    for (int mt = 0; mt < 2; ++mt) { s[mt][0] = zv; s[mt][1] = zv; }
    #pragma unroll
    for (int nt = 0; nt < 2; ++nt) {
      #pragma unroll
      for (int kk = 0; kk < 2; ++kk) {
        Frag kb;
        const unsigned short* krow =
            K + (size_t)(b * L_ + k0 + nt * 16 + m) * D_ + dsl + kk * 32 + hi * 8;
        kb.u[0] = ((const v4u*)krow)[0];
        kb.u[1] = ((const v4u*)krow)[2];
        #pragma unroll
        for (int mt = 0; mt < 2; ++mt)
          s[mt][nt] = wmma_bf16(qf[mt][kk], kb, s[mt][nt]);
      }
    }
    #pragma unroll
    for (int mt = 0; mt < 2; ++mt)
      #pragma unroll
      for (int nt = 0; nt < 2; ++nt)
        #pragma unroll
        for (int r = 0; r < 8; ++r)
          Sp[w][mt * 16 + hi * 8 + r][nt * 16 + m] = s[mt][nt][r];
    __syncthreads();

    // split-K reduce + online softmax: 8 threads per row, shuffle reductions
    {
      float sv[4];
      #pragma unroll
      for (int j = 0; j < 4; ++j) {
        float acc = 0.f;
        #pragma unroll
        for (int ww = 0; ww < 8; ++ww) acc += Sp[ww][row8][c8 + 8 * j];
        sv[j] = acc;
      }
      float tmax = fmaxf(fmaxf(sv[0], sv[1]), fmaxf(sv[2], sv[3]));
      tmax = fmaxf(tmax, __shfl_xor(tmax, 1, 32));
      tmax = fmaxf(tmax, __shfl_xor(tmax, 2, 32));
      tmax = fmaxf(tmax, __shfl_xor(tmax, 4, 32));
      float mold = mrun[row8];
      float mnew = fmaxf(mold, tmax);
      float psum = 0.f;
      #pragma unroll
      for (int j = 0; j < 4; ++j) {
        float pv = __expf(sv[j] - mnew);
        P[row8][c8 + 8 * j] = f2bf(pv);
        psum += pv;
      }
      psum += __shfl_xor(psum, 1, 32);
      psum += __shfl_xor(psum, 2, 32);
      psum += __shfl_xor(psum, 4, 32);
      if (c8 == 0) {
        float sc = __expf(mold - mnew);
        lrun[row8] = lrun[row8] * sc + psum;
        mrun[row8] = mnew;
        rscale[row8] = sc;
      }
    }
    __syncthreads();

    // O = O*rscale + P @ V_slice
    Frag pf[2];
    float scl[2][8];
    #pragma unroll
    for (int mt = 0; mt < 2; ++mt) {
      pf[mt].u[0] = *(const v4u*)&P[mt * 16 + m][hi * 8];
      pf[mt].u[1] = *(const v4u*)&P[mt * 16 + m][16 + hi * 8];
      #pragma unroll
      for (int r = 0; r < 8; ++r) scl[mt][r] = rscale[mt * 16 + hi * 8 + r];
    }
    #pragma unroll
    for (int nt = 0; nt < 4; ++nt) {
      Frag vf;
      const unsigned short* vrow =
          Vt + ((size_t)b * D_ + dsl + nt * 16 + m) * L_ + k0 + hi * 8;
      vf.u[0] = ((const v4u*)vrow)[0];
      vf.u[1] = ((const v4u*)vrow)[2];
      #pragma unroll
      for (int mt = 0; mt < 2; ++mt) {
        #pragma unroll
        for (int r = 0; r < 8; ++r) o[mt][nt][r] *= scl[mt][r];
        o[mt][nt] = wmma_bf16(pf[mt], vf, o[mt][nt]);
      }
    }
    __syncthreads();
  }

  // epilogue: normalize and store fp32
  #pragma unroll
  for (int mt = 0; mt < 2; ++mt) {
    float inv[8];
    #pragma unroll
    for (int r = 0; r < 8; ++r) inv[r] = 1.0f / lrun[mt * 16 + hi * 8 + r];
    #pragma unroll
    for (int nt = 0; nt < 4; ++nt)
      #pragma unroll
      for (int r = 0; r < 8; ++r)
        Out[(size_t)(b * L_ + m0 + mt * 16 + hi * 8 + r) * D_ + dsl + nt * 16 + m] =
            o[mt][nt][r] * inv[r];
  }
}

extern "C" void kernel_launch(void* const* d_in, const int* in_sizes, int n_in,
                              void* d_out, int out_size, void* d_ws, size_t ws_size,
                              hipStream_t stream) {
  const float* raw  = (const float*)d_in[0];
  const float* hand = (const float*)d_in[1];
  const float* Wq_r = (const float*)d_in[2];  const float* bq_r = (const float*)d_in[3];
  const float* Wk_r = (const float*)d_in[4];  const float* bk_r = (const float*)d_in[5];
  const float* Wv_r = (const float*)d_in[6];  const float* bv_r = (const float*)d_in[7];
  const float* Wq_h = (const float*)d_in[8];  const float* bq_h = (const float*)d_in[9];
  const float* Wk_h = (const float*)d_in[10]; const float* bk_h = (const float*)d_in[11];
  const float* Wv_h = (const float*)d_in[12]; const float* bv_h = (const float*)d_in[13];

  const size_t XSZ = (size_t)BL_ * D_;
  unsigned short* ws  = (unsigned short*)d_ws;
  unsigned short* xr  = ws;
  unsigned short* xh  = xr + XSZ;
  unsigned short* wt  = xh + XSZ;
  unsigned short* qr  = wt + (size_t)6 * D_ * D_;
  unsigned short* kr  = qr + XSZ;
  unsigned short* qh  = kr + XSZ;
  unsigned short* kh  = qh + XSZ;
  unsigned short* vtr = kh + XSZ;
  unsigned short* vth = vtr + XSZ;

  cvt_inputs<<<(int)(XSZ / 256), 256, 0, stream>>>(raw, hand, xr, xh);
  cvt_weights<<<6 * D_ * D_ / 256, 256, 0, stream>>>(Wq_r, Wk_r, Wv_r, Wq_h, Wk_h, Wv_h, wt);

  // 6 proj * 8 N-blocks * 128 M-chunks
  proj_kernel<<<6 * 8 * 128, 256, 0, stream>>>(xr, xh, wt,
                                               bq_r, bk_r, bv_r, bq_h, bk_h, bv_h,
                                               qr, kr, qh, kh, vtr, vth);

  dim3 ag(L_ / 32, B_, 2);
  attn_kernel<<<ag, 256, 0, stream>>>(qr, kh, vtr, qh, kr, vth, (float*)d_out);
}
